// MultiHeadAttention_59554016526846
// MI455X (gfx1250) — matrix operand order repaired and checked by reading, not compiled, not run on hardware
//
#include <hip/hip_runtime.h>

// ---------------------------------------------------------------------------
// MI455X (gfx1250) fused multi-head self-attention, bf16 WMMA + fp32 accum.
//   B=2, L=2048, DIM=1024, H=16, D=64.  SCALE = 64^-0.25 applied to q and k.
// Round 3: fix async-to-LDS builtin pointer types (v4i AS1* / v4i AS3*).
// ---------------------------------------------------------------------------

typedef __attribute__((ext_vector_type(16))) __bf16 v16bf;
typedef __attribute__((ext_vector_type(8)))  float  v8f;

#define DIMC   1024
#define LSEQ   2048
#define NHEAD  16
#define HDIM   64
#define NBATCH 2
#define MTOK   (NBATCH * LSEQ)       // 4096 tokens
#define SCALE_F 0.35355339059327373f // 64^-0.25

#if __has_builtin(__builtin_amdgcn_global_load_async_to_lds_b128) && \
    __has_builtin(__builtin_amdgcn_s_wait_asynccnt)
#define HAVE_ASYNC_LDS 1
#else
#define HAVE_ASYNC_LDS 0
#endif

// --- WMMA wrapper -----------------------------------------------------------
__device__ __forceinline__ v8f wmma_bf16(v16bf a, v16bf b, v8f c) {
  return __builtin_amdgcn_wmma_f32_16x16x32_bf16(
      /*neg_a=*/false, a, /*neg_b=*/false, b,
      /*c_mod=*/(short)0, c, /*reuse_a=*/false, /*reuse_b=*/false);
}

__device__ __forceinline__ v16bf frag_ld2(const __bf16* p0, const __bf16* p1) {
  union { uint4 u[2]; v16bf v; } f;
  f.u[0] = *reinterpret_cast<const uint4*>(p0);
  f.u[1] = *reinterpret_cast<const uint4*>(p1);
  return f.v;
}

// A fragment 16x32 (row-major source, ld elems):
//   lane l: row = l&15; K chunks (l>>4)*8 .. +8  and  16+(l>>4)*8 .. +8
__device__ __forceinline__ v16bf load_frag_a(const __bf16* base, int ld, int lane) {
  const int m  = lane & 15;
  const int kc = (lane >> 4) << 3;
  const __bf16* r = base + (size_t)m * ld;
  return frag_ld2(r + kc, r + 16 + kc);
}

// B fragment 32x16 read from B^T layout [N][K] (row-major, ld elems):
//   lane l: col n = l&15; K chunks (l>>4)*8 .. +8  and  16+(l>>4)*8 .. +8
__device__ __forceinline__ v16bf load_frag_b(const __bf16* base, int ld, int lane) {
  const int n  = lane & 15;
  const int kc = (lane >> 4) << 3;
  const __bf16* r = base + (size_t)n * ld + kc;
  return frag_ld2(r, r + 16);
}

#if HAVE_ASYNC_LDS
// GLOBAL_LOAD_ASYNC_TO_LDS_B128: per-lane 16B global -> LDS, ASYNCcnt-tracked.
// Builtin signature (from hipcc diagnostic): param is pointer to
// int __attribute__((vector_size(16))) in the global (AS1) / LDS (AS3) spaces.
typedef int v4i_vs __attribute__((vector_size(16)));
typedef __attribute__((address_space(1))) v4i_vs* gptr_v4i;
typedef __attribute__((address_space(3))) v4i_vs* lptr_v4i;

__device__ __forceinline__ void async_copy_b128(const void* g, void* l) {
  __builtin_amdgcn_global_load_async_to_lds_b128(
      (gptr_v4i)g, (lptr_v4i)l, 0, 0);
}
#endif

// 16-lane (half-wave) row reductions for softmax; rows of a C-tile live on a
// fixed 16-lane group, so xor masks 1,2,4,8 stay inside the group.
__device__ __forceinline__ float rmax16(float v) {
  v = fmaxf(v, __shfl_xor(v, 1, 32));
  v = fmaxf(v, __shfl_xor(v, 2, 32));
  v = fmaxf(v, __shfl_xor(v, 4, 32));
  v = fmaxf(v, __shfl_xor(v, 8, 32));
  return v;
}
__device__ __forceinline__ float rsum16(float v) {
  v += __shfl_xor(v, 1, 32);
  v += __shfl_xor(v, 2, 32);
  v += __shfl_xor(v, 4, 32);
  v += __shfl_xor(v, 8, 32);
  return v;
}

// ---------------------------------------------------------------------------
// x (fp32) -> bf16, 4 elements / thread
// ---------------------------------------------------------------------------
__global__ void __launch_bounds__(256) convert_x_bf16(
    const float* __restrict__ x, __bf16* __restrict__ xb) {
  const size_t i = ((size_t)blockIdx.x * 256 + threadIdx.x) * 4;
  float4 f = *reinterpret_cast<const float4*>(x + i);
  union { __bf16 h[4]; unsigned long long u; } o;
  o.h[0] = (__bf16)f.x; o.h[1] = (__bf16)f.y;
  o.h[2] = (__bf16)f.z; o.h[3] = (__bf16)f.w;
  *reinterpret_cast<unsigned long long*>(xb + i) = o.u;
}

// ---------------------------------------------------------------------------
// W[k][n] fp32 -> W^T[n][k] bf16, LDS-tiled 32x32 (z selects which weight)
// ---------------------------------------------------------------------------
__global__ void __launch_bounds__(256) transpose_w_bf16(
    const float* __restrict__ w0, const float* __restrict__ w1,
    const float* __restrict__ w2, const float* __restrict__ w3,
    __bf16* __restrict__ wt_base) {
  const float* srcs[4] = {w0, w1, w2, w3};
  const float* __restrict__ src = srcs[blockIdx.z];
  __bf16* __restrict__ dst = wt_base + (size_t)blockIdx.z * DIMC * DIMC;

  __shared__ float tile[32][33];
  const int n0 = blockIdx.x * 32;
  const int k0 = blockIdx.y * 32;
  const int tx = threadIdx.x;   // 0..31
  const int ty = threadIdx.y;   // 0..7
#pragma unroll
  for (int r = 0; r < 32; r += 8)
    tile[ty + r][tx] = src[(size_t)(k0 + ty + r) * DIMC + n0 + tx];
  __syncthreads();
#pragma unroll
  for (int r = 0; r < 32; r += 8)
    dst[(size_t)(n0 + ty + r) * DIMC + k0 + tx] = (__bf16)tile[tx][ty + r];
}

// ---------------------------------------------------------------------------
// GEMM: C[4096x1024] = A[4096x1024](bf16) * B (given as B^T[1024x1024] bf16)
// 256 thr (8 waves). Block tile 128(M)x64(N); wave = 16 rows x 64 cols.
// B^T tile (64x32) double-buffered in LDS (stride 40 -> conflict-free b128);
// async global->LDS copy when available; ONE barrier per K-step.
// mode 0: dst=bf16 q/k  [B,H,L,D], value * SCALE
// mode 1: dst=bf16 v^T  [B,H,D,L]
// mode 2: dst=fp32 out  [M,N], += bias[n]
// ---------------------------------------------------------------------------
__global__ void __launch_bounds__(256) gemm_bf16_wmma(
    const __bf16* __restrict__ A, const __bf16* __restrict__ BT,
    void* __restrict__ dst, const float* __restrict__ bias, int mode) {
  __shared__ alignas(16) __bf16 ldsb[2][64 * 40];

  const int lane = threadIdx.x & 31;
  const int warp = threadIdx.x >> 5;
  const int n0   = blockIdx.x * 64;
  const int mrow = blockIdx.y * 128 + warp * 16;

  v8f acc[4] = { {}, {}, {}, {} };

  const int tld_n = threadIdx.x >> 2;        // 0..63
  const int tld_k = (threadIdx.x & 3) << 3;  // 0,8,16,24
  const __bf16* gB = &BT[(size_t)(n0 + tld_n) * DIMC + tld_k];
  __bf16* lB0 = &ldsb[0][tld_n * 40 + tld_k];
  __bf16* lB1 = &ldsb[1][tld_n * 40 + tld_k];
  const __bf16* gA = A + (size_t)mrow * DIMC;

  constexpr int NK = DIMC / 32;  // 32 K-steps

  v16bf afrag = load_frag_a(gA, DIMC, lane);

#if HAVE_ASYNC_LDS
  async_copy_b128(gB, lB0);
  for (int kt = 0; kt < NK; ++kt) {
    __builtin_amdgcn_s_wait_asynccnt(0);   // tile kt has landed in LDS
    __syncthreads();                       // visible; prior readers drained
    if (kt + 1 < NK)                       // overlap tile kt+1 with compute
      async_copy_b128(gB + (size_t)(kt + 1) * 32, (kt & 1) ? lB0 : lB1);
    const __bf16* lbase = &ldsb[kt & 1][0];
#else
  uint4 pref = *reinterpret_cast<const uint4*>(gB);
  for (int kt = 0; kt < NK; ++kt) {
    *reinterpret_cast<uint4*>((kt & 1) ? lB1 : lB0) = pref;
    if (kt + 1 < NK)
      pref = *reinterpret_cast<const uint4*>(gB + (size_t)(kt + 1) * 32);
    __syncthreads();
    const __bf16* lbase = &ldsb[kt & 1][0];
#endif
    // batch all fragment loads, then issue WMMAs back-to-back
    v16bf bf[4];
#pragma unroll
    for (int t = 0; t < 4; ++t)
      bf[t] = load_frag_b(lbase + t * 16 * 40, 40, lane);
    v16bf anext;
    if (kt + 1 < NK)
      anext = load_frag_a(gA + (size_t)(kt + 1) * 32, DIMC, lane);
#pragma unroll
    for (int t = 0; t < 4; ++t)
      acc[t] = wmma_bf16(afrag, bf[t], acc[t]);
    if (kt + 1 < NK) afrag = anext;
  }

  // epilogue: C layout => VGPR r, lane l -> row r+8*(l>>4), col l&15
#pragma unroll
  for (int t = 0; t < 4; ++t) {
#pragma unroll
    for (int r = 0; r < 8; ++r) {
      const int row = mrow + r + 8 * (lane >> 4);
      const int col = n0 + t * 16 + (lane & 15);
      const float v = acc[t][r];
      if (mode == 2) {
        ((float*)dst)[(size_t)row * DIMC + col] = v + bias[col];
      } else {
        const int bb = row >> 11, ii = row & (LSEQ - 1);
        const int hh = col >> 6,  dd = col & (HDIM - 1);
        const size_t head = (size_t)bb * NHEAD + hh;
        if (mode == 0)
          ((__bf16*)dst)[(head * LSEQ + ii) * HDIM + dd] = (__bf16)(v * SCALE_F);
        else
          ((__bf16*)dst)[(head * HDIM + dd) * LSEQ + ii] = (__bf16)v;
      }
    }
  }
}

// ---------------------------------------------------------------------------
// Flash attention: one wave owns a 16-query tile of one (b,h).
//   q,k: [B,H,L,64] bf16 (pre-scaled); vT: [B,H,64,L] bf16
//   out attn: [B*L, 1024] bf16 (token-major, ready for the Wo GEMM)
// Per 32-key chunk: K/V fragments loaded up-front (latency hides under the
// softmax VALU/exp work), 4 WMMA (S), online softmax, P via per-wave LDS
// layout conversion (C-layout -> A-fragment), 4 WMMA (P*V).
// ---------------------------------------------------------------------------
__global__ void __launch_bounds__(256) flash_attn_wmma(
    const __bf16* __restrict__ q, const __bf16* __restrict__ k,
    const __bf16* __restrict__ vT, __bf16* __restrict__ attn) {
  __shared__ alignas(16) __bf16 p_lds_all[8 * 16 * 40];

  const int lane = threadIdx.x & 31;
  const int warp = threadIdx.x >> 5;
  __bf16* p_lds = p_lds_all + warp * 16 * 40;

  const int wid = blockIdx.x * 8 + warp;       // 0..4095
  const int i0  = (wid & 127) * 16;            // query tile base
  const int hh  = (wid >> 7) & (NHEAD - 1);
  const int bb  = wid >> 11;

  const size_t head = (size_t)bb * NHEAD + hh;
  const __bf16* qbase = q  + (head * LSEQ + i0) * HDIM;
  const __bf16* kbase = k  +  head * LSEQ * HDIM;
  const __bf16* vbase = vT +  head * HDIM * LSEQ;

  // Q fragments: resident for the whole KV sweep (d 0..32 and 32..64)
  const v16bf A0 = load_frag_a(qbase,      HDIM, lane);
  const v16bf A1 = load_frag_a(qbase + 32, HDIM, lane);

  v8f acc[4] = { {}, {}, {}, {} };
  float rowmax[8], rowsum[8];
#pragma unroll
  for (int r = 0; r < 8; ++r) { rowmax[r] = -1e30f; rowsum[r] = 0.0f; }

  for (int j0 = 0; j0 < LSEQ; j0 += 32) {
    // ---- issue ALL global loads for this chunk first -------------------
    v16bf vb[4];
#pragma unroll
    for (int t = 0; t < 4; ++t)
      vb[t] = load_frag_b(vbase + (size_t)(t * 16) * LSEQ + j0, LSEQ, lane);
    const __bf16* kb0 = kbase + (size_t)j0 * HDIM;
    const __bf16* kb1 = kb0 + 16 * HDIM;
    v16bf kf[4];
    kf[0] = load_frag_b(kb0,      HDIM, lane);
    kf[1] = load_frag_b(kb0 + 32, HDIM, lane);
    kf[2] = load_frag_b(kb1,      HDIM, lane);
    kf[3] = load_frag_b(kb1 + 32, HDIM, lane);

    // ---- S = (q*s)(k*s)^T for two 16-key tiles -------------------------
    v8f z0 = {}, z1 = {};
    z0 = wmma_bf16(A0, kf[0], z0);
    z0 = wmma_bf16(A1, kf[1], z0);
    z1 = wmma_bf16(A0, kf[2], z1);
    z1 = wmma_bf16(A1, kf[3], z1);

    // ---- online softmax over the 32-key chunk --------------------------
    float corr[8];
#pragma unroll
    for (int r = 0; r < 8; ++r) {
      const float s0 = z0[r], s1 = z1[r];
      const float mr   = rmax16(fmaxf(s0, s1));
      const float newm = fmaxf(rowmax[r], mr);
      corr[r] = __expf(rowmax[r] - newm);
      rowmax[r] = newm;
      const float p0 = __expf(s0 - newm);
      const float p1 = __expf(s1 - newm);
      rowsum[r] = rowsum[r] * corr[r] + rsum16(p0 + p1);
      const int row = r + 8 * (lane >> 4);
      p_lds[row * 40 + (lane & 15)]      = (__bf16)p0;
      p_lds[row * 40 + 16 + (lane & 15)] = (__bf16)p1;
    }
    // rescale running O
#pragma unroll
    for (int t = 0; t < 4; ++t)
#pragma unroll
      for (int r = 0; r < 8; ++r) acc[t][r] *= corr[r];

    // same-wave LDS ops are in-order in HW; stop compiler reordering only
    asm volatile("" ::: "memory");

    // P: C-layout -> A-fragment (16x32) via LDS
    const v16bf PA = load_frag_a(p_lds, 40, lane);

    // ---- O += P * V (V^T layout makes B fragments contiguous) ----------
#pragma unroll
    for (int t = 0; t < 4; ++t)
      acc[t] = wmma_bf16(PA, vb[t], acc[t]);
  }

  // ---- normalize and emit [B*L, DIM] bf16 ------------------------------
  float inv[8];
#pragma unroll
  for (int r = 0; r < 8; ++r) inv[r] = 1.0f / rowsum[r];
#pragma unroll
  for (int t = 0; t < 4; ++t)
#pragma unroll
    for (int r = 0; r < 8; ++r) {
      const int row = r + 8 * (lane >> 4);
      const size_t tok = (size_t)bb * LSEQ + i0 + row;
      attn[tok * DIMC + hh * HDIM + t * 16 + (lane & 15)] =
          (__bf16)(acc[t][r] * inv[r]);
    }
}

// ---------------------------------------------------------------------------
// Launch: convert -> transpose weights -> QKV GEMMs -> flash attn -> O GEMM
// Workspace: 48 MiB of bf16 staging.
// ---------------------------------------------------------------------------
extern "C" void kernel_launch(void* const* d_in, const int* in_sizes, int n_in,
                              void* d_out, int out_size, void* d_ws, size_t ws_size,
                              hipStream_t stream) {
  const float* x  = (const float*)d_in[0];
  const float* Wq = (const float*)d_in[1];
  const float* Wk = (const float*)d_in[2];
  const float* Wv = (const float*)d_in[3];
  const float* Wo = (const float*)d_in[4];
  const float* bo = (const float*)d_in[5];

  char* ws = (char*)d_ws;
  const size_t MB = 1024 * 1024;
  __bf16* xb   = (__bf16*)(ws + 0 * MB);    // 8 MiB: x bf16 [4096,1024]
  __bf16* wt   = (__bf16*)(ws + 8 * MB);    // 8 MiB: Wq^T,Wk^T,Wv^T,Wo^T bf16
  __bf16* qs   = (__bf16*)(ws + 16 * MB);   // 8 MiB: q*SCALE [B,H,L,64]
  __bf16* kscl = (__bf16*)(ws + 24 * MB);   // 8 MiB: k*SCALE [B,H,L,64]
  __bf16* vTm  = (__bf16*)(ws + 32 * MB);   // 8 MiB: v^T     [B,H,64,L]
  __bf16* attn = (__bf16*)(ws + 40 * MB);   // 8 MiB: attn out [B*L,1024]

  __bf16* wtq = wt;
  __bf16* wtk = wt + (size_t)DIMC * DIMC;
  __bf16* wtv = wt + (size_t)2 * DIMC * DIMC;
  __bf16* wto = wt + (size_t)3 * DIMC * DIMC;

  convert_x_bf16<<<(MTOK * DIMC) / (256 * 4), 256, 0, stream>>>(x, xb);
  transpose_w_bf16<<<dim3(32, 32, 4), dim3(32, 8), 0, stream>>>(Wq, Wk, Wv, Wo, wt);

  const dim3 ggrid(DIMC / 64, MTOK / 128);  // (16, 32)
  gemm_bf16_wmma<<<ggrid, 256, 0, stream>>>(xb, wtq, qs,   nullptr, 0);
  gemm_bf16_wmma<<<ggrid, 256, 0, stream>>>(xb, wtk, kscl, nullptr, 0);
  gemm_bf16_wmma<<<ggrid, 256, 0, stream>>>(xb, wtv, vTm,  nullptr, 1);

  flash_attn_wmma<<<(NBATCH * NHEAD * (LSEQ / 16)) / 8, 256, 0, stream>>>(
      qs, kscl, vTm, attn);

  gemm_bf16_wmma<<<ggrid, 256, 0, stream>>>(attn, wto, d_out, bo, 2);
}
